// GraphRNNBilinear_9131100471763
// MI455X (gfx1250) — compile-verified, matrix operand order checked by reading, not run on hardware
//
#include <hip/hip_runtime.h>
#include <hip/hip_bf16.h>
#include <math.h>

// Problem constants (from reference): B=8, S=128, M=128, D=256, H=256, T=256
#define BB 8
#define SS 128
#define MM 128
#define DD 256
#define HH 256
#define TT 256
#define NROWS (BB * SS)   // 1024 flattened rows

typedef __bf16 bf16_t;
typedef bf16_t v16bf __attribute__((ext_vector_type(16)));
typedef float  v8f   __attribute__((ext_vector_type(8)));

static __device__ __forceinline__ bf16_t f2bf(float f) {
  unsigned u = __builtin_bit_cast(unsigned, f);
  u += 0x7fffu + ((u >> 16) & 1u);               // round-to-nearest-even
  return __builtin_bit_cast(bf16_t, (unsigned short)(u >> 16));
}
static __device__ __forceinline__ float bf2f(bf16_t b) {
  unsigned u = ((unsigned)__builtin_bit_cast(unsigned short, b)) << 16;
  return __builtin_bit_cast(float, u);
}
static __device__ __forceinline__ v8f vzero8() {
  v8f z = {0.f, 0.f, 0.f, 0.f, 0.f, 0.f, 0.f, 0.f};
  return z;
}
static __device__ __forceinline__ v8f wmma_bf16(v16bf a, v16bf b, v8f c) {
  // D = A(16x32) * B(32x16) + C, fp32 accumulate
  return __builtin_amdgcn_wmma_f32_16x16x32_bf16(false, a, false, b, (short)0, c,
                                                 false, false);
}

// Load one 16x32 bf16 fragment (A-operand layout; B uses the mirrored pattern
// with N striped over lanes, loaded from row-major W[N][K]).
// tile points at element [0][0] of the 16x32 region; ld is row stride in elems.
static __device__ __forceinline__ v16bf frag_bf16(const bf16_t* tile, int ld) {
  const int lane = threadIdx.x & 31;
  const bf16_t* p = tile + (lane & 15) * ld + ((lane >> 4) << 3);
  v16bf v;
#pragma unroll
  for (int i = 0; i < 8; ++i) v[i] = p[i];          // K = base .. base+7
#pragma unroll
  for (int i = 0; i < 8; ++i) v[i + 8] = p[i + 16]; // K = base+16 .. base+23
  return v;
}

// ---------------------------------------------------------------------------
// fp32 -> bf16 conversion
__global__ void cvt_f32_bf16(const float* __restrict__ s, bf16_t* __restrict__ d,
                             int n) {
  int i = blockIdx.x * blockDim.x + threadIdx.x;
  if (i < n) d[i] = f2bf(s[i]);
}

// ---------------------------------------------------------------------------
// out(Mx256) = act( A(MxK,bf16) @ W(256xK,bf16)^T + bias0 (+bias1) )
// grid.x = M/16 row tiles; 512 threads = 16 waves, wave w owns N-tile w.
template <int ACT> // 0 = none, 1 = ELU
__global__ __launch_bounds__(512, 1) void gemm_bias_act(
    const bf16_t* __restrict__ A, const bf16_t* __restrict__ W,
    const float* __restrict__ bias0, const float* __restrict__ bias1,
    float* __restrict__ out, int K) {
  const int m0 = blockIdx.x * 16;
  const int lane = threadIdx.x & 31;
  const int w = threadIdx.x >> 5;
  const int n0 = w * 16;
  v8f acc = vzero8();
  for (int kc = 0; kc < K; kc += 32) {
    v16bf a = frag_bf16(A + m0 * K + kc, K);
    v16bf b = frag_bf16(W + n0 * K + kc, K);
    acc = wmma_bf16(a, b, acc);
  }
  const int n = n0 + (lane & 15);
  float bv = bias0[n] + (bias1 ? bias1[n] : 0.f);
  const int rbase = (lane >> 4) * 8;
#pragma unroll
  for (int r = 0; r < 8; ++r) {
    float v = acc[r] + bv;
    if (ACT == 1) v = v > 0.f ? v : (expf(v) - 1.f);
    out[(m0 + rbase + r) * 256 + n] = v;
  }
}

// ---------------------------------------------------------------------------
// Graph RNN scan. Single workgroup, 512 threads (16 waves), 128 steps.
// h state (batch 8, padded to 16 rows) kept in bf16 LDS; weights streamed
// from L2-resident bf16 copies. Xp = input@gW_ih0^T + gb_ih0 + gb_hh0
// precomputed. Emits graph_states (masked y) as bf16.
#define GLD 264
__global__ __launch_bounds__(512, 1) void graph_scan(
    const float* __restrict__ Xp,          // (B*S, H) fp32, row = b*S + t
    const int* __restrict__ mask,          // (B, S)
    const bf16_t* __restrict__ Whh0, const bf16_t* __restrict__ Wih1,
    const bf16_t* __restrict__ Whh1, const float* __restrict__ gb_ih1,
    const float* __restrict__ gb_hh1,
    bf16_t* __restrict__ gsb)              // (B*S, H) bf16 graph_states
{
  __shared__ bf16_t h0s[16 * GLD];
  __shared__ bf16_t h1s[16 * GLD];
  __shared__ int lens[BB];

  const int tid = threadIdx.x;
  const int lane = tid & 31;
  const int w = tid >> 5;
  const int n0 = w * 16;
  const int cn = n0 + (lane & 15);
  const int rbase = (lane >> 4) * 8;
  const float b1c = gb_ih1[cn] + gb_hh1[cn];

  if (tid < BB) {
    int s = 0;
    for (int i = 0; i < SS; ++i) s += mask[tid * SS + i];
    lens[tid] = s;
  }
  for (int e = tid; e < 16 * GLD; e += 512) { h0s[e] = f2bf(0.f); h1s[e] = f2bf(0.f); }
  __syncthreads();

  for (int t = 0; t < SS; ++t) {
    // h0' = tanh(Xp[:,t] + h0 @ gW_hh0^T), masked
    v8f acc = vzero8();
#pragma unroll
    for (int kc = 0; kc < 8; ++kc) {
      v16bf a = frag_bf16(h0s + kc * 32, GLD);
      v16bf b = frag_bf16(Whh0 + n0 * HH + kc * 32, HH);
      acc = wmma_bf16(a, b, acc);
    }
    __syncthreads();
    if (rbase == 0) { // only rows 0..7 are real batch rows
#pragma unroll
      for (int r = 0; r < 8; ++r) {
        bf16_t oldv = h0s[r * GLD + cn];
        float nv = tanhf(Xp[((r << 7) + t) * HH + cn] + acc[r]);
        h0s[r * GLD + cn] = (t < lens[r]) ? f2bf(nv) : oldv;
      }
    }
    __syncthreads();
    // h1' = tanh(h0' @ gW_ih1^T + h1 @ gW_hh1^T + b), masked; y = v ? h1 : 0
    v8f acc1 = vzero8();
#pragma unroll
    for (int kc = 0; kc < 8; ++kc) {
      v16bf a = frag_bf16(h0s + kc * 32, GLD);
      v16bf b = frag_bf16(Wih1 + n0 * HH + kc * 32, HH);
      acc1 = wmma_bf16(a, b, acc1);
    }
#pragma unroll
    for (int kc = 0; kc < 8; ++kc) {
      v16bf a = frag_bf16(h1s + kc * 32, GLD);
      v16bf b = frag_bf16(Whh1 + n0 * HH + kc * 32, HH);
      acc1 = wmma_bf16(a, b, acc1);
    }
    __syncthreads();
    if (rbase == 0) {
#pragma unroll
      for (int r = 0; r < 8; ++r) {
        bool v = (t < lens[r]);
        bf16_t oldv = h1s[r * GLD + cn];
        float nv = tanhf(acc1[r] + b1c);
        bf16_t h = v ? f2bf(nv) : oldv;
        h1s[r * GLD + cn] = h;
        gsb[((r << 7) + t) * HH + cn] = v ? f2bf(nv) : f2bf(0.f);
      }
    }
    __syncthreads();
  }
}

// ---------------------------------------------------------------------------
// Edge RNN scan: row-independent, so 64 workgroups each own 16 rows and run
// the 128-step recurrence locally. The A@eW_ih0^T term is maintained as a
// fp32 rank-1 running update ("contrib"); the three HxH GEMMs use WMMA with
// bf16 state in LDS and weights streamed from L2.
#define ELD 264
#define CLD 260
__global__ __launch_bounds__(512, 1) void edge_scan(
    const bf16_t* __restrict__ gsb,        // (1024, H) bf16 initial h0=h1
    const bf16_t* __restrict__ Whh0, const bf16_t* __restrict__ Wih1,
    const bf16_t* __restrict__ Whh1,
    const float* __restrict__ Wih0f,       // eW_ih0 (H, M) fp32
    const float* __restrict__ eb_ih0, const float* __restrict__ eb_hh0,
    const float* __restrict__ eb_ih1, const float* __restrict__ eb_hh1,
    const float* __restrict__ clsW, const float* __restrict__ clsb,
    float* __restrict__ Aout)              // (1024, M) fp32
{
  __shared__ bf16_t sb0[16 * ELD], sb1[16 * ELD], sb2[16 * ELD], sb3[16 * ELD];
  __shared__ float contrib[16 * CLD];
  __shared__ float Arow[16 * MM];
  __shared__ float edgev[16];

  const int tid = threadIdx.x;
  const int lane = tid & 31;
  const int w = tid >> 5;                  // wave id == owned row for reductions
  const int n0 = w * 16;
  const int cn = n0 + (lane & 15);
  const int rbase = (lane >> 4) * 8;
  const int row0 = blockIdx.x * 16;

  const float ebc0 = eb_ih0[cn] + eb_hh0[cn];
  const float ebc1 = eb_ih1[cn] + eb_hh1[cn];
  const float cb = clsb[0];

  for (int e = tid; e < 16 * HH; e += 512) {
    int r = e >> 8, c = e & 255;
    bf16_t v = gsb[(row0 + r) * HH + c];
    sb0[r * ELD + c] = v;
    sb1[r * ELD + c] = v;
    contrib[r * CLD + c] = 0.f;
  }
  for (int e = tid; e < 16 * MM; e += 512) Arow[e] = 0.f;
  __syncthreads();

  bf16_t* h0 = sb0; bf16_t* h1 = sb1; bf16_t* e0 = sb2; bf16_t* e1 = sb3;

  for (int j = 0; j < MM; ++j) {
    // e0 = tanh(contrib + h0 @ eW_hh0^T + b)
    v8f acc = vzero8();
#pragma unroll
    for (int kc = 0; kc < 8; ++kc) {
      v16bf a = frag_bf16(h0 + kc * 32, ELD);
      v16bf b = frag_bf16(Whh0 + n0 * HH + kc * 32, HH);
      acc = wmma_bf16(a, b, acc);
    }
    __syncthreads();
#pragma unroll
    for (int r = 0; r < 8; ++r) {
      int row = rbase + r;
      float v = tanhf(contrib[row * CLD + cn] + acc[r] + ebc0);
      e0[row * ELD + cn] = f2bf(v);
    }
    __syncthreads();
    // e1 = tanh(e0 @ eW_ih1^T + h1 @ eW_hh1^T + b)
    v8f acc1 = vzero8();
#pragma unroll
    for (int kc = 0; kc < 8; ++kc) {
      v16bf a = frag_bf16(e0 + kc * 32, ELD);
      v16bf b = frag_bf16(Wih1 + n0 * HH + kc * 32, HH);
      acc1 = wmma_bf16(a, b, acc1);
    }
#pragma unroll
    for (int kc = 0; kc < 8; ++kc) {
      v16bf a = frag_bf16(h1 + kc * 32, ELD);
      v16bf b = frag_bf16(Whh1 + n0 * HH + kc * 32, HH);
      acc1 = wmma_bf16(a, b, acc1);
    }
    __syncthreads();
#pragma unroll
    for (int r = 0; r < 8; ++r) {
      int row = rbase + r;
      e1[row * ELD + cn] = f2bf(tanhf(acc1[r] + ebc1));
    }
    __syncthreads();
    // edge = sigmoid(e1 . cls_W + cls_b); wave w reduces row w
    {
      float p = 0.f;
#pragma unroll
      for (int i = 0; i < 8; ++i) {
        int c = lane + 32 * i;
        p += bf2f(e1[w * ELD + c]) * clsW[c];
      }
#pragma unroll
      for (int off = 16; off > 0; off >>= 1) p += __shfl_down(p, off, 32);
      if (lane == 0) {
        float s = 1.f / (1.f + expf(-(p + cb)));
        edgev[w] = s;
        Arow[w * MM + j] = s;
      }
    }
    __syncthreads();
    // rank-1 update: contrib[:, c] += edge * eW_ih0[c, j]
    {
      float eg = edgev[w];
#pragma unroll
      for (int i = 0; i < 8; ++i) {
        int c = lane + 32 * i;
        contrib[w * CLD + c] += eg * Wih0f[c * MM + j];
      }
    }
    __syncthreads();
    bf16_t* t0 = h0; h0 = e0; e0 = t0;
    bf16_t* t1 = h1; h1 = e1; e1 = t1;
  }
  for (int e = tid; e < 16 * MM; e += 512) Aout[row0 * MM + e] = Arow[e];
}

// ---------------------------------------------------------------------------
// arc_logits[b][m][s] = (s > m) ? A[b][s][m] : 0
__global__ void arc_finalize(const float* __restrict__ A, float* __restrict__ out,
                             int n) {
  int i = blockIdx.x * blockDim.x + threadIdx.x;
  if (i >= n) return;
  int b = i >> 14;
  int m = (i >> 7) & 127;
  int s = i & 127;
  out[i] = (s > m) ? A[(b * SS + s) * MM + m] : 0.f;
}

// ---------------------------------------------------------------------------
extern "C" void kernel_launch(void* const* d_in, const int* in_sizes, int n_in,
                              void* d_out, int out_size, void* d_ws,
                              size_t ws_size, hipStream_t stream) {
  (void)in_sizes; (void)n_in; (void)out_size; (void)ws_size;
  const float* input   = (const float*)d_in[0];
  const int*   mask    = (const int*)d_in[2];
  const float* gW_ih0  = (const float*)d_in[5];
  const float* gW_hh0  = (const float*)d_in[6];
  const float* gb_ih0  = (const float*)d_in[7];
  const float* gb_hh0  = (const float*)d_in[8];
  const float* gW_ih1  = (const float*)d_in[9];
  const float* gW_hh1  = (const float*)d_in[10];
  const float* gb_ih1  = (const float*)d_in[11];
  const float* gb_hh1  = (const float*)d_in[12];
  const float* eW_ih0  = (const float*)d_in[13];
  const float* eW_hh0  = (const float*)d_in[14];
  const float* eb_ih0  = (const float*)d_in[15];
  const float* eb_hh0  = (const float*)d_in[16];
  const float* eW_ih1  = (const float*)d_in[17];
  const float* eW_hh1  = (const float*)d_in[18];
  const float* eb_ih1  = (const float*)d_in[19];
  const float* eb_hh1  = (const float*)d_in[20];
  const float* cls_W   = (const float*)d_in[21];
  const float* cls_b   = (const float*)d_in[22];
  const float* head_W  = (const float*)d_in[23];
  const float* head_b  = (const float*)d_in[24];
  const float* dep_W   = (const float*)d_in[25];
  const float* dep_b   = (const float*)d_in[26];

  float* out = (float*)d_out;
  char* ws = (char*)d_ws;
  size_t off = 0;
  auto alloc = [&](size_t bytes) {
    void* p = ws + off;
    off += (bytes + 255) & ~(size_t)255;
    return p;
  };
  bf16_t* Xbf     = (bf16_t*)alloc(NROWS * DD * sizeof(bf16_t));
  bf16_t* gWih0b  = (bf16_t*)alloc(HH * DD * sizeof(bf16_t));
  bf16_t* gWhh0b  = (bf16_t*)alloc(HH * HH * sizeof(bf16_t));
  bf16_t* gWih1b  = (bf16_t*)alloc(HH * HH * sizeof(bf16_t));
  bf16_t* gWhh1b  = (bf16_t*)alloc(HH * HH * sizeof(bf16_t));
  bf16_t* eWhh0b  = (bf16_t*)alloc(HH * HH * sizeof(bf16_t));
  bf16_t* eWih1b  = (bf16_t*)alloc(HH * HH * sizeof(bf16_t));
  bf16_t* eWhh1b  = (bf16_t*)alloc(HH * HH * sizeof(bf16_t));
  bf16_t* headWb  = (bf16_t*)alloc(TT * HH * sizeof(bf16_t));
  bf16_t* depWb   = (bf16_t*)alloc(TT * HH * sizeof(bf16_t));
  bf16_t* gsb     = (bf16_t*)alloc(NROWS * HH * sizeof(bf16_t));
  float*  Xp      = (float*)alloc(NROWS * HH * sizeof(float));
  float*  Aflat   = (float*)alloc(NROWS * MM * sizeof(float));

  auto cvt = [&](const float* s, bf16_t* d, int n) {
    cvt_f32_bf16<<<(n + 511) / 512, 512, 0, stream>>>(s, d, n);
  };
  cvt(input,  Xbf,    NROWS * DD);
  cvt(gW_ih0, gWih0b, HH * DD);
  cvt(gW_hh0, gWhh0b, HH * HH);
  cvt(gW_ih1, gWih1b, HH * HH);
  cvt(gW_hh1, gWhh1b, HH * HH);
  cvt(eW_hh0, eWhh0b, HH * HH);
  cvt(eW_ih1, eWih1b, HH * HH);
  cvt(eW_hh1, eWhh1b, HH * HH);
  cvt(head_W, headWb, TT * HH);
  cvt(dep_W,  depWb,  TT * HH);

  // Xp = input @ gW_ih0^T + gb_ih0 + gb_hh0  (hoisted from the scan)
  gemm_bias_act<0><<<NROWS / 16, 512, 0, stream>>>(Xbf, gWih0b, gb_ih0, gb_hh0,
                                                   Xp, DD);
  // sequential graph RNN scan -> graph_states (bf16)
  graph_scan<<<1, 512, 0, stream>>>(Xp, mask, gWhh0b, gWih1b, gWhh1b, gb_ih1,
                                    gb_hh1, gsb);
  // head/dep tags: ELU(gs @ W^T + b)
  gemm_bias_act<1><<<NROWS / 16, 512, 0, stream>>>(gsb, headWb, head_b, nullptr,
                                                   out, HH);
  gemm_bias_act<1><<<NROWS / 16, 512, 0, stream>>>(gsb, depWb, dep_b, nullptr,
                                                   out + NROWS * TT, HH);
  // edge RNN: 64 row-groups, each scans 128 steps independently
  edge_scan<<<NROWS / 16, 512, 0, stream>>>(gsb, eWhh0b, eWih1b, eWhh1b, eW_ih0,
                                            eb_ih0, eb_hh0, eb_ih1, eb_hh1,
                                            cls_W, cls_b, Aflat);
  // tril mask + transpose
  arc_finalize<<<(BB * MM * SS + 511) / 512, 512, 0, stream>>>(
      Aflat, out + 2 * NROWS * TT, BB * MM * SS);
}
